// MLPExtractor_51848845197843
// MI455X (gfx1250) — matrix operand order repaired and checked by reading, not compiled, not run on hardware
//
#include <hip/hip_runtime.h>
#include <hip/hip_bf16.h>
#include <math.h>

// ---------------------------------------------------------------------------
// Problem constants (match reference)
// ---------------------------------------------------------------------------
#define BB    32
#define NN    100
#define HH    256
#define HAA   1024
#define HCC   512
#define KACT  5000
#define NP1   101            // N+1 rows in embedded_features
#define EMBW  356            // H + N columns
#define ROWS  32             // rows per block (2 M-tiles)
#define RT    157            // ceil(5000/32) row tiles

#define USE_ASYNC_LDS 1      // gfx1250 GLOBAL_LOAD_ASYNC_TO_LDS path

typedef _Float16 half_t;
typedef __attribute__((ext_vector_type(16))) _Float16 v16h;
typedef __attribute__((ext_vector_type(8)))  _Float16 v8h;
typedef __attribute__((ext_vector_type(8)))  float    v8f;

#define LDX 776              // 768 + 8 halfs pad (bank spread for b128 reads)
#define LDH 1032             // 1024 + 8 halfs pad

// branchless tanh: (e-1)/(e+1), e = exp(2*clamp(x,±15)); native exp + rcp div
__device__ __forceinline__ float tanh_fast(float x) {
  float xc = fminf(fmaxf(x, -15.0f), 15.0f);
  float e  = __expf(2.0f * xc);
  return __fdividef(e - 1.0f, e + 1.0f);
}

// ---------------------------------------------------------------------------
// 0. zero the scatter target region of d_out
// ---------------------------------------------------------------------------
__global__ void zero_out_kernel(float* __restrict__ out, int n) {
  int i = blockIdx.x * blockDim.x + threadIdx.x;
  if (i < n) out[i] = 0.0f;
}

// ---------------------------------------------------------------------------
// 1. stream-compact mask -> per-batch ascending index lists (matches top_k)
// ---------------------------------------------------------------------------
__global__ __launch_bounds__(256) void compact_idx_kernel(
    const float* __restrict__ emb, int* __restrict__ idxbuf) {
  __shared__ int wsum[8];
  __shared__ int wbase[8];
  __shared__ int runbase;
  const int b = blockIdx.x;
  const int t = threadIdx.x;
  const int lane = t & 31, wv = t >> 5;
  if (t == 0) runbase = 0;
  __syncthreads();
  for (int c = 0; c < 40; ++c) {
    int f = c * 256 + t;                 // flat pair index in [0,10000)
    bool pred = false;
    if (f < NN * NN) {
      int i = f / NN, j = f % NN;
      float v = emb[(size_t)b * NP1 * EMBW + (size_t)(1 + i) * EMBW + HH + j];
      pred = (v > 0.5f);
    }
    unsigned mb = (unsigned)__ballot(pred);
    if (lane == 0) wsum[wv] = __popc(mb);
    __syncthreads();
    if (t == 0) {
      int s = runbase;
      for (int w = 0; w < 8; ++w) { wbase[w] = s; s += wsum[w]; }
      runbase = s;
    }
    __syncthreads();
    if (pred) {
      int pos = wbase[wv] + __popc(mb & ((1u << lane) - 1));
      idxbuf[b * KACT + pos] = f;
    }
    __syncthreads();
  }
}

// ---------------------------------------------------------------------------
// 2. convert node features (first 256 cols of embedded) to f16
// ---------------------------------------------------------------------------
__global__ void cvt_gn_kernel(const float* __restrict__ emb,
                              half_t* __restrict__ gn_h, int n) {
  int e = blockIdx.x * blockDim.x + threadIdx.x;
  if (e >= n) return;
  int b    = e / (NP1 * HH);
  int rem  = e % (NP1 * HH);
  int node = rem / HH;
  int c    = rem % HH;
  gn_h[e] = (half_t)emb[(size_t)b * NP1 * EMBW + (size_t)node * EMBW + c];
}

// ---------------------------------------------------------------------------
// 3. convert weights into WMMA-B-fragment-linear f16 layout:
//    frag = nt*KS + ks ; within frag: lane (0..31) x 16 halfs contiguous
//    hb = lane>>4 : h[j] = W[ks*32 + (j<8 ? 8*hb + j : 16 + 8*hb + (j-8))][nt*16 + (lane&15)]
// ---------------------------------------------------------------------------
__global__ void cvt_wfrag_kernel(const float* __restrict__ W,
                                 half_t* __restrict__ Wf,
                                 int KS, int NOUT, int n) {
  int e = blockIdx.x * blockDim.x + threadIdx.x;
  if (e >= n) return;
  int j    = e & 15;
  int lc   = e >> 4;        // lane-chunk index
  int lane = lc & 31;
  int frag = lc >> 5;
  int ks   = frag % KS;
  int nt   = frag / KS;
  int K    = ks * 32 + ((j < 8) ? ((lane >> 4) * 8 + j) : (16 + (lane >> 4) * 8 + (j - 8)));
  int ncol = nt * 16 + (lane & 15);
  Wf[e] = (half_t)W[(size_t)K * NOUT + ncol];
}

// ---------------------------------------------------------------------------
// helper: async gather of one 16-byte chunk into LDS (ASYNCcnt-tracked)
// ---------------------------------------------------------------------------
__device__ __forceinline__ void ld_async_b128(void* lds_dst, const void* gsrc) {
#if USE_ASYNC_LDS
  unsigned dst32 = (unsigned)(size_t)lds_dst;     // LDS aperture: addr[31:0]
  unsigned long long src64 = (unsigned long long)(size_t)gsrc;
  asm volatile("global_load_async_to_lds_b128 %0, %1, off"
               :: "v"(dst32), "v"(src64) : "memory");
#else
  *(uint4*)lds_dst = *(const uint4*)gsrc;
#endif
}

__device__ __forceinline__ void wait_async_lds() {
#if USE_ASYNC_LDS
  asm volatile("s_wait_asynccnt 0x0" ::: "memory");
#endif
}

// ---------------------------------------------------------------------------
// 4. fused pair-MLP: [g|n1|n2](32x768) -> tanh(1024) -> tanh(1024) -> logit
//    grid = (157 row-tiles, 32 batches), 256 threads = 8 waves
//    wave owns 8 N-tiles, processed as 2 groups of (2 M-tiles x 4 N-tiles)
// ---------------------------------------------------------------------------
__global__ __launch_bounds__(256) void pair_mlp_kernel(
    const half_t* __restrict__ gn_h, const int* __restrict__ idxbuf,
    const v16h* __restrict__ w1f, const float* __restrict__ ab1,
    const v16h* __restrict__ w2f, const float* __restrict__ ab2,
    const float* __restrict__ aW3, const float* __restrict__ ab3,
    float* __restrict__ logits) {
  __shared__ half_t sX[ROWS * LDX];     // 49664 B
  __shared__ half_t sH1[ROWS * LDH];    // 66048 B
  __shared__ float  sLogit[ROWS];
  __shared__ int    sI1[ROWS], sI2[ROWS];

  const int b    = blockIdx.y;
  const int tile = blockIdx.x;
  const int t    = threadIdx.x;
  const int lane = t & 31;
  const int wv   = t >> 5;

  if (t < ROWS) {
    int kk = tile * ROWS + t;
    if (kk > KACT - 1) kk = KACT - 1;          // clamp padded rows
    int p = idxbuf[b * KACT + kk];
    sI1[t] = p / NN;
    sI2[t] = p % NN;
  } else if (t < 2 * ROWS) {
    sLogit[t - ROWS] = 0.0f;
  }
  __syncthreads();

  // gather X tile: 32 rows x 768 halfs = 3072 16B-chunks, 12 per thread
  for (int i = 0; i < 12; ++i) {
    int e8  = t + i * 256;
    int r   = e8 / 96;
    int c8  = e8 % 96;
    int sec = c8 >> 5;                         // 0:g 1:n1 2:n2
    int w8  = c8 & 31;
    int node = (sec == 0) ? 0 : ((sec == 1) ? 1 + sI1[r] : 1 + sI2[r]);
    ld_async_b128(&sX[r * LDX + c8 * 8],
                  gn_h + (size_t)(b * NP1 + node) * HH + w8 * 8);
  }
  wait_async_lds();
  __syncthreads();

  const int M0   = lane & 15;                  // A-fragment row within tile
  const int kb   = (lane >> 4) * 8;            // A-fragment K base (0 / 8)
  const int mrow = (lane >> 4) * 8;            // C-fragment row base
  const int ncol = lane & 15;                  // C-fragment column in tile

  // ---------------- layer 1: X(32x768) @ aW1 -> tanh -> sH1 ----------------
  for (int g = 0; g < 2; ++g) {
    v8f acc[2][4] = {};
    for (int ks = 0; ks < 24; ++ks) {
      v16h a[2];
      for (int m = 0; m < 2; ++m) {
        int row = m * 16 + M0;
        v8h lo = *(const v8h*)(&sX[row * LDX + ks * 32 + kb]);
        v8h hi = *(const v8h*)(&sX[row * LDX + ks * 32 + kb + 16]);
        for (int j = 0; j < 8; ++j) { a[m][j] = lo[j]; a[m][8 + j] = hi[j]; }
      }
      for (int n = 0; n < 4; ++n) {
        int nt = wv * 8 + g * 4 + n;
        v16h bf = w1f[(nt * 24 + ks) * 32 + lane];
        for (int m = 0; m < 2; ++m) {
          acc[m][n] = __builtin_amdgcn_wmma_f32_16x16x32_f16(
              false, a[m], false, bf, (short)0, acc[m][n], false, false);
        }
      }
    }
    for (int n = 0; n < 4; ++n) {
      int nn = (wv * 8 + g * 4 + n) * 16 + ncol;
      float bias = ab1[nn];
      for (int m = 0; m < 2; ++m) {
        for (int r = 0; r < 8; ++r) {
          sH1[(m * 16 + mrow + r) * LDH + nn] =
              (half_t)tanh_fast(acc[m][n][r] + bias);
        }
      }
    }
  }
  __syncthreads();

  // ------------- layer 2: sH1(32x1024) @ aW2 -> tanh -> dot aW3 -----------
  float pl[16];
  for (int r = 0; r < 16; ++r) pl[r] = 0.0f;
  for (int g = 0; g < 2; ++g) {
    v8f acc[2][4] = {};
    for (int ks = 0; ks < 32; ++ks) {
      v16h a[2];
      for (int m = 0; m < 2; ++m) {
        int row = m * 16 + M0;
        v8h lo = *(const v8h*)(&sH1[row * LDH + ks * 32 + kb]);
        v8h hi = *(const v8h*)(&sH1[row * LDH + ks * 32 + kb + 16]);
        for (int j = 0; j < 8; ++j) { a[m][j] = lo[j]; a[m][8 + j] = hi[j]; }
      }
      for (int n = 0; n < 4; ++n) {
        int nt = wv * 8 + g * 4 + n;
        v16h bf = w2f[(nt * 32 + ks) * 32 + lane];
        for (int m = 0; m < 2; ++m) {
          acc[m][n] = __builtin_amdgcn_wmma_f32_16x16x32_f16(
              false, a[m], false, bf, (short)0, acc[m][n], false, false);
        }
      }
    }
    for (int n = 0; n < 4; ++n) {
      int nn = (wv * 8 + g * 4 + n) * 16 + ncol;
      float bias = ab2[nn];
      float w3   = aW3[nn];
      for (int m = 0; m < 2; ++m) {
        for (int r = 0; r < 8; ++r) {
          pl[m * 8 + r] += tanh_fast(acc[m][n][r] + bias) * w3;
        }
      }
    }
  }
  for (int m = 0; m < 2; ++m) {
    for (int r = 0; r < 8; ++r) {
      atomicAdd(&sLogit[m * 16 + mrow + r], pl[m * 8 + r]);
    }
  }
  __syncthreads();

  if (t < ROWS) {
    int kk = tile * ROWS + t;
    if (kk < KACT) logits[b * KACT + kk] = sLogit[t] + ab3[0];
  }
}

// ---------------------------------------------------------------------------
// 5. softmax over 5000 logits + scatter into (B,10000) grid
// ---------------------------------------------------------------------------
__global__ __launch_bounds__(256) void softmax_scatter_kernel(
    const float* __restrict__ logits, const int* __restrict__ idxbuf,
    float* __restrict__ out) {
  __shared__ float red[256];
  __shared__ float smax, ssum;
  const int b = blockIdx.x;
  const int t = threadIdx.x;

  float m = -1e30f;
  for (int k = t; k < KACT; k += 256) m = fmaxf(m, logits[b * KACT + k]);
  red[t] = m;
  __syncthreads();
  for (int s = 128; s > 0; s >>= 1) {
    if (t < s) red[t] = fmaxf(red[t], red[t + s]);
    __syncthreads();
  }
  if (t == 0) smax = red[0];
  __syncthreads();

  float acc = 0.0f;
  for (int k = t; k < KACT; k += 256)
    acc += __expf(logits[b * KACT + k] - smax);
  red[t] = acc;
  __syncthreads();
  for (int s = 128; s > 0; s >>= 1) {
    if (t < s) red[t] += red[t + s];
    __syncthreads();
  }
  if (t == 0) ssum = red[0];
  __syncthreads();

  float inv = 1.0f / ssum;
  for (int k = t; k < KACT; k += 256) {
    int f = idxbuf[b * KACT + k];
    out[(size_t)b * NN * NN + f] = __expf(logits[b * KACT + k] - smax) * inv;
  }
}

// ---------------------------------------------------------------------------
// 6. value head (tiny, scalar f32): one block per batch
// ---------------------------------------------------------------------------
__global__ __launch_bounds__(256) void value_head_kernel(
    const float* __restrict__ emb,
    const float* __restrict__ cW1, const float* __restrict__ cb1,
    const float* __restrict__ cW2, const float* __restrict__ cb2,
    const float* __restrict__ cW3, const float* __restrict__ cb3,
    float* __restrict__ out) {
  __shared__ float sg[HH];
  __shared__ float h1[HCC];
  __shared__ float h2[HCC];
  __shared__ float red[256];
  const int b = blockIdx.x;
  const int t = threadIdx.x;

  if (t < HH) sg[t] = emb[(size_t)b * NP1 * EMBW + t];   // g row (node 0)
  __syncthreads();

  for (int col = t; col < HCC; col += 256) {
    float a = cb1[col];
    for (int k = 0; k < HH; ++k) a += sg[k] * cW1[(size_t)k * HCC + col];
    h1[col] = tanh_fast(a);
  }
  __syncthreads();
  for (int col = t; col < HCC; col += 256) {
    float a = cb2[col];
    for (int k = 0; k < HCC; ++k) a += h1[k] * cW2[(size_t)k * HCC + col];
    h2[col] = tanh_fast(a);
  }
  __syncthreads();
  float acc = 0.0f;
  for (int k = t; k < HCC; k += 256) acc += h2[k] * cW3[k];
  red[t] = acc;
  __syncthreads();
  for (int s = 128; s > 0; s >>= 1) {
    if (t < s) red[t] += red[t + s];
    __syncthreads();
  }
  if (t == 0) out[(size_t)BB * NN * NN + b] = red[0] + cb3[0];
}

// ---------------------------------------------------------------------------
// launch
// ---------------------------------------------------------------------------
extern "C" void kernel_launch(void* const* d_in, const int* in_sizes, int n_in,
                              void* d_out, int out_size, void* d_ws,
                              size_t ws_size, hipStream_t stream) {
  const float* emb = (const float*)d_in[0];
  const float* aW1 = (const float*)d_in[1];
  const float* ab1 = (const float*)d_in[2];
  const float* aW2 = (const float*)d_in[3];
  const float* ab2 = (const float*)d_in[4];
  const float* aW3 = (const float*)d_in[5];
  const float* ab3 = (const float*)d_in[6];
  const float* cW1 = (const float*)d_in[7];
  const float* cb1 = (const float*)d_in[8];
  const float* cW2 = (const float*)d_in[9];
  const float* cb2 = (const float*)d_in[10];
  const float* cW3 = (const float*)d_in[11];
  const float* cb3 = (const float*)d_in[12];
  float* out = (float*)d_out;

  // workspace carve (all offsets multiples of 256B)
  char* ws = (char*)d_ws;
  int*    idxbuf = (int*)ws;      ws += (size_t)BB * KACT * 4;        // 640000
  half_t* gn_h   = (half_t*)ws;   ws += (size_t)BB * NP1 * HH * 2;    // 1654784
  half_t* w1h    = (half_t*)ws;   ws += (size_t)(3 * HH) * HAA * 2;   // 1572864
  half_t* w2h    = (half_t*)ws;   ws += (size_t)HAA * HAA * 2;        // 2097152
  float*  logits = (float*)ws;    ws += (size_t)BB * KACT * 4;        // 640000

  // 0: zero scatter grid
  zero_out_kernel<<<(BB * NN * NN + 255) / 256, 256, 0, stream>>>(
      out, BB * NN * NN);
  // 1: index compaction
  compact_idx_kernel<<<BB, 256, 0, stream>>>(emb, idxbuf);
  // 2: feature conversion
  {
    int n = BB * NP1 * HH;
    cvt_gn_kernel<<<(n + 255) / 256, 256, 0, stream>>>(emb, gn_h, n);
  }
  // 3: weight fragment conversion
  {
    int n1 = 3 * HH * HAA;  // 786432
    cvt_wfrag_kernel<<<(n1 + 255) / 256, 256, 0, stream>>>(aW1, w1h, 24, HAA, n1);
    int n2 = HAA * HAA;     // 1048576
    cvt_wfrag_kernel<<<(n2 + 255) / 256, 256, 0, stream>>>(aW2, w2h, 32, HAA, n2);
  }
  // 4: fused pair MLP (WMMA)
  pair_mlp_kernel<<<dim3(RT, BB), 256, 0, stream>>>(
      gn_h, idxbuf, (const v16h*)w1h, ab1, (const v16h*)w2h, ab2, aW3, ab3,
      logits);
  // 5: softmax + scatter
  softmax_scatter_kernel<<<BB, 256, 0, stream>>>(logits, idxbuf, out);
  // 6: value head
  value_head_kernel<<<BB, 256, 0, stream>>>(emb, cW1, cb1, cW2, cb2, cW3, cb3,
                                            out);
}
